// MiCRoLayer_90228672954959
// MI455X (gfx1250) — compile-verified, matrix operand order checked by reading, not run on hardware
//
#include <hip/hip_runtime.h>

#define HDIM 1024
#define FDIM 4096
#define NTOK 16384   // B*L = 8*2048
#define NEXP 4

typedef __attribute__((ext_vector_type(16))) __bf16 v16bf;
typedef __attribute__((ext_vector_type(8)))  float  v8f;

// ---------------------------------------------------------------------------
// Convert fp32 weights [E][K][N] (row-major) into bf16 WMMA B-fragment order:
// dst[e][nb][kb][lane][16], nb over N/16, kb over K/32.
// Per-lane packing follows ISA 16-bit operand layout:
//   lane = h*16 + n%16 ; element idx = v*2+r ; K = kb*32 + kpat(v,h,r)
//   kpat(v,h,r) = (v<4 ? 2v : 16+2(v-4)) + 8h + r
// ---------------------------------------------------------------------------
__global__ __launch_bounds__(256) void pack_w(const float* __restrict__ src,
                                              __bf16* __restrict__ dst,
                                              int K, int N, int KBLK) {
    long tid  = (long)blockIdx.x * 256 + threadIdx.x;   // over E*K*N
    long perE = (long)K * N;
    int  e    = (int)(tid / perE);
    long r    = tid - (long)e * perE;
    int  fi   = (int)(r >> 9);       // fragment index within expert
    int  r512 = (int)(r & 511);
    int  ln   = r512 >> 4;           // lane 0..31
    int  idx  = r512 & 15;           // packed element 0..15
    int  nb   = fi / KBLK;
    int  kb   = fi - nb * KBLK;
    int  n    = nb * 16 + (ln & 15);
    int  h    = ln >> 4;
    int  v    = idx >> 1;
    int  rr   = idx & 1;
    int  kk   = (v < 4 ? 2 * v : 16 + 2 * (v - 4)) + 8 * h + rr;
    int  k    = kb * 32 + kk;
    dst[tid] = (__bf16)src[((long)e * K + k) * N + n];
}

// ---------------------------------------------------------------------------
// Router: one wave per token. logits[e] = x[t] . Wr[:,e] + br[e]; top-1.
// ---------------------------------------------------------------------------
__global__ __launch_bounds__(256) void router_k(const float* __restrict__ x,
                                                const float* __restrict__ Wr,
                                                const float* __restrict__ br,
                                                int* __restrict__ top) {
    const int lane = threadIdx.x & 31;
    const int wave = threadIdx.x >> 5;
    const int t    = blockIdx.x * 8 + wave;
    const float* xr = x + (size_t)t * HDIM;
    float a0 = 0.f, a1 = 0.f, a2 = 0.f, a3 = 0.f;
    #pragma unroll 4
    for (int j = 0; j < HDIM / 32; ++j) {
        int   k  = j * 32 + lane;
        float xv = xr[k];
        float4 w = ((const float4*)Wr)[k];   // Wr is [H][4] row-major
        a0 += xv * w.x; a1 += xv * w.y; a2 += xv * w.z; a3 += xv * w.w;
    }
    for (int off = 16; off; off >>= 1) {
        a0 += __shfl_xor(a0, off, 32);
        a1 += __shfl_xor(a1, off, 32);
        a2 += __shfl_xor(a2, off, 32);
        a3 += __shfl_xor(a3, off, 32);
    }
    if (lane == 0) {
        float l0 = a0 + br[0], l1 = a1 + br[1], l2 = a2 + br[2], l3 = a3 + br[3];
        int best = 0; float bv = l0;
        if (l1 > bv) { bv = l1; best = 1; }   // strict > keeps first max (jnp.argmax)
        if (l2 > bv) { bv = l2; best = 2; }
        if (l3 > bv) { bv = l3; best = 3; }
        top[t] = best;
    }
}

__global__ void zero_counts(int* __restrict__ c) {
    if (threadIdx.x < NEXP) c[threadIdx.x] = 0;
}

__global__ __launch_bounds__(256) void scatter_k(const int* __restrict__ top,
                                                 int* __restrict__ counts,
                                                 int* __restrict__ idx_list) {
    int t = blockIdx.x * 256 + threadIdx.x;
    int e = top[t];
    int p = atomicAdd(&counts[e], 1);
    idx_list[e * NTOK + p] = t;
}

// ---------------------------------------------------------------------------
// Fused expert FFN: block = 16 tokens of one expert, 256 threads = 8 waves.
//   Xs : X tile  [16 x 1024] bf16, stored as 32 A-fragments (kb)  -> 32 KB
//   H1s: H1 tile [16 x 4096] bf16, stored as 128 A-fragments      -> 128 KB
// GEMM1: H1 = relu(X @ We1[e] + be1[e]); GEMM2: out = H1 @ We2[e] + be2[e]
// Each wave processes TWO adjacent N-tiles per A-fragment load: two
// independent WMMA accumulation chains + half the LDS read traffic.
// ---------------------------------------------------------------------------
__global__ __launch_bounds__(256) void ffn_kernel(const float* __restrict__ x,
                                                  const float* __restrict__ be1,
                                                  const float* __restrict__ be2,
                                                  const __bf16* __restrict__ W1s,
                                                  const __bf16* __restrict__ W2s,
                                                  const int* __restrict__ counts,
                                                  const int* __restrict__ idx_list,
                                                  float* __restrict__ out) {
    __shared__ v16bf Xs[32 * 32];     //  32 KB
    __shared__ v16bf H1s[128 * 32];   // 128 KB  (total 160 KB -> 2 blocks/WGP)

    const int e     = blockIdx.y;
    const int cnt   = counts[e];
    const int start = blockIdx.x * 16;
    if (start >= cnt) return;
    const int valid = (cnt - start < 16) ? (cnt - start) : 16;
    const int* tlist = idx_list + e * NTOK + start;

    const int tid  = threadIdx.x;
    const int lane = tid & 31;
    const int wave = tid >> 5;

    // ---- Stage X tile -> LDS (f32 -> bf16, swizzled into A-fragment order)
    {
        __bf16* xsb = (__bf16*)Xs;
        #pragma unroll 4
        for (int i = 0; i < 64; ++i) {
            int s = tid + i * 256;           // 0..16383
            int m = s >> 10;                 // token row 0..15
            int k = s & 1023;
            float val = 0.f;
            if (m < valid) val = x[(size_t)tlist[m] * HDIM + k];
            int kb = k >> 5, kk = k & 31;
            int h  = (kk >> 3) & 1;
            int v  = (kk >> 4) * 4 + ((kk & 7) >> 1);
            int r  = kk & 1;
            int ln = h * 16 + m;
            xsb[((kb * 32 + ln) << 4) + v * 2 + r] = (__bf16)val;
        }
    }
    __syncthreads();

    // ---- GEMM1: each wave owns 512 F columns = 16 pairs of N-tiles; K = 1024
    {
        const v16bf* W1e = (const v16bf*)W1s + (size_t)e * 256 * 32 * 32;
        __bf16* h1b = (__bf16*)H1s;
        for (int nt = 0; nt < 16; ++nt) {
            int nb0 = wave * 32 + nt * 2;          // even
            const v16bf* bf0 = W1e + ((size_t)nb0 * 32) * 32 + lane;
            const v16bf* bf1 = bf0 + 32 * 32;      // nb0 + 1
            v8f c0 = 0.f, c1 = 0.f;
            #pragma unroll 4
            for (int kb = 0; kb < 32; ++kb) {
                v16bf a  = Xs[kb * 32 + lane];
                v16bf b0 = bf0[kb * 32];
                v16bf b1 = bf1[kb * 32];
                c0 = __builtin_amdgcn_wmma_f32_16x16x32_bf16(
                         false, a, false, b0, (short)0, c0, false, false);
                c1 = __builtin_amdgcn_wmma_f32_16x16x32_bf16(
                         false, a, false, b1, (short)0, c1, false, false);
            }
            // bias + relu, write into H1 A-fragment layout.
            // nb0 even, nb1 = nb0+1 -> both land in fragment kb2 = nb0>>1.
            int   n     = lane & 15;
            float bias0 = be1[e * FDIM + nb0 * 16 + n];
            float bias1 = be1[e * FDIM + nb0 * 16 + 16 + n];
            int   kb2   = nb0 >> 1;
            int   vlo   = ((n & 7) >> 1) * 2 + (n & 1);   // v*2+r for nb even
            int   hp    = (n >> 3) & 1;
            int   mbase = (lane >> 4) * 8;
            #pragma unroll
            for (int i2 = 0; i2 < 8; ++i2) {
                int m = mbase + i2;
                int base = ((kb2 * 32 + hp * 16 + m) << 4) + vlo;
                float h0 = fmaxf(c0[i2] + bias0, 0.f);
                float h1 = fmaxf(c1[i2] + bias1, 0.f);
                h1b[base]     = (__bf16)h0;       // nb even -> v' in 0..3
                h1b[base + 8] = (__bf16)h1;       // nb odd  -> v' in 4..7
            }
        }
    }
    __syncthreads();

    // ---- GEMM2: each wave owns 128 H columns = 4 pairs of N-tiles; K = 4096
    {
        const v16bf* W2e = (const v16bf*)W2s + (size_t)e * 64 * 128 * 32;
        for (int t = 0; t < 4; ++t) {
            int nb0 = wave * 8 + t * 2;
            const v16bf* bf0 = W2e + ((size_t)nb0 * 128) * 32 + lane;
            const v16bf* bf1 = bf0 + 128 * 32;     // nb0 + 1
            v8f c0 = 0.f, c1 = 0.f;
            #pragma unroll 4
            for (int kb = 0; kb < 128; ++kb) {
                v16bf a  = H1s[kb * 32 + lane];
                v16bf b0 = bf0[kb * 32];
                v16bf b1 = bf1[kb * 32];
                c0 = __builtin_amdgcn_wmma_f32_16x16x32_bf16(
                         false, a, false, b0, (short)0, c0, false, false);
                c1 = __builtin_amdgcn_wmma_f32_16x16x32_bf16(
                         false, a, false, b1, (short)0, c1, false, false);
            }
            int   n     = lane & 15;
            int   h0col = nb0 * 16 + n;
            int   h1col = h0col + 16;
            float bias0 = be2[e * HDIM + h0col];
            float bias1 = be2[e * HDIM + h1col];
            int   mbase = (lane >> 4) * 8;
            #pragma unroll
            for (int i2 = 0; i2 < 8; ++i2) {
                int m = mbase + i2;
                if (m < valid) {
                    float* orow = out + (size_t)tlist[m] * HDIM;
                    orow[h0col] = c0[i2] + bias0;
                    orow[h1col] = c1[i2] + bias1;
                }
            }
        }
    }
}

// ---------------------------------------------------------------------------
extern "C" void kernel_launch(void* const* d_in, const int* in_sizes, int n_in,
                              void* d_out, int out_size, void* d_ws, size_t ws_size,
                              hipStream_t stream) {
    const float* x   = (const float*)d_in[0];
    const float* Wr  = (const float*)d_in[1];
    const float* br  = (const float*)d_in[2];
    const float* We1 = (const float*)d_in[3];
    const float* be1 = (const float*)d_in[4];
    const float* We2 = (const float*)d_in[5];
    const float* be2 = (const float*)d_in[6];
    float* out = (float*)d_out;

    char* ws = (char*)d_ws;
    int*    counts = (int*)ws;                                     // 16 B (pad 256)
    int*    top    = (int*)(ws + 256);                             // NTOK ints
    int*    idx    = (int*)(ws + 256 + 4 * NTOK);                  // NEXP*NTOK ints
    __bf16* W1s    = (__bf16*)(ws + 256 + 4 * NTOK + 4 * NEXP * NTOK);
    __bf16* W2s    = W1s + (size_t)NEXP * HDIM * FDIM;
    // ws usage: 256 + 64K + 256K + 32M + 32M ~= 67.4 MB

    // 1) pre-convert + fragment-swizzle weights (fp32 -> bf16)
    pack_w<<<(NEXP * HDIM * FDIM) / 256, 256, 0, stream>>>(We1, W1s, HDIM, FDIM, HDIM / 32);
    pack_w<<<(NEXP * FDIM * HDIM) / 256, 256, 0, stream>>>(We2, W2s, FDIM, HDIM, FDIM / 32);

    // 2) router (top-1 per token)
    router_k<<<NTOK / 8, 256, 0, stream>>>(x, Wr, br, top);

    // 3) compact tokens per expert
    zero_counts<<<1, 32, 0, stream>>>(counts);
    scatter_k<<<NTOK / 256, 256, 0, stream>>>(top, counts, idx);

    // 4) fused expert FFN (WMMA bf16)
    dim3 grid(NTOK / 16, NEXP);
    ffn_kernel<<<grid, 256, 0, stream>>>(x, be1, be2, W1s, W2s, counts, idx, out);
}
